// MoE_49890340110395
// MI455X (gfx1250) — matrix operand order repaired and checked by reading, not compiled, not run on hardware
//
#include <hip/hip_runtime.h>
#include <hip/hip_bf16.h>

// ---------------------------------------------------------------------------
// Sparse top-2 MoE for MI455X (gfx1250, wave32, WMMA).
// T=1024, D=1024, F=2048, E=16, TOP_K=2.
// Memory-bound: 402MB of f32 weights @ 23.3 TB/s ~= 17us floor; bf16 WMMA
// (v_wmma_f32_16x16x32_bf16) keeps the ~26 GFLOP of sparse compute under it.
// Weight slabs are software-pipelined through registers (global_load_b128 for
// slab K+32 issued while WMMAs for slab K run), converted f32->bf16 with one
// v_cvt_pk_bf16_f32 per packed word (__builtin_convertvector), and staged
// pair-packed (two K-adjacent bf16 per u32) in LDS matching the WMMA operand
// layouts (ISA 7.12.2).
// ---------------------------------------------------------------------------

typedef __bf16 bf16_t;
typedef __attribute__((ext_vector_type(16))) __bf16 v16bf;
typedef __attribute__((ext_vector_type(2)))  __bf16 v2bf;
typedef __attribute__((ext_vector_type(8)))  float  v8f;
typedef __attribute__((ext_vector_type(4)))  float  f32x4;
typedef __attribute__((ext_vector_type(2)))  float  f32x2;
typedef __attribute__((ext_vector_type(4)))  int    i32x4;
typedef __attribute__((ext_vector_type(8)))  int    i32x8;

#define T_TOK 1024
#define DDIM  1024
#define FDIM  2048
#define EDIM  16
#define MAX_RT 64   // max 16-row tiles per expert (<= T rows per expert)

// f32 pair -> one u32 of two bf16 (lo->[15:0]); lowers to v_cvt_pk_bf16_f32
__device__ __forceinline__ unsigned int pack2(float lo, float hi) {
  f32x2 f; f[0] = lo; f[1] = hi;
  v2bf b = __builtin_convertvector(f, v2bf);
  return __builtin_bit_cast(unsigned int, b);
}

// ---------------------------------------------------------------------------
// 1. Router: logits -> softmax -> top-2 (renormalized), count tokens/expert.
// ---------------------------------------------------------------------------
__global__ void moe_router(const float* __restrict__ x,
                           const float* __restrict__ Wr,      // D x E
                           float* __restrict__ topk_w,        // T*2
                           int*   __restrict__ topk_idx,      // T*2
                           int*   __restrict__ counts)        // E
{
  const int t   = blockIdx.x;
  const int tid = threadIdx.x;
  const int e   = tid & 15;
  const int c   = tid >> 4;            // 16 chunks of 64 along D
  const float* xr = x + (size_t)t * DDIM;

  float p = 0.f;
  #pragma unroll 4
  for (int d = c * 64; d < c * 64 + 64; ++d)
    p += xr[d] * Wr[d * EDIM + e];

  __shared__ float part[256];
  __shared__ float logit[EDIM];
  part[tid] = p;
  __syncthreads();
  if (tid < EDIM) {
    float s = 0.f;
    #pragma unroll
    for (int c2 = 0; c2 < 16; ++c2) s += part[c2 * 16 + tid];
    logit[tid] = s;
  }
  __syncthreads();
  if (tid == 0) {
    float mx = logit[0];
    #pragma unroll
    for (int i = 1; i < EDIM; ++i) mx = fmaxf(mx, logit[i]);
    float pr[EDIM];
    #pragma unroll
    for (int i = 0; i < EDIM; ++i) pr[i] = __expf(logit[i] - mx);
    int i0 = 0;
    for (int i = 1; i < EDIM; ++i) if (pr[i] > pr[i0]) i0 = i;
    int i1 = (i0 == 0) ? 1 : 0;
    for (int i = 0; i < EDIM; ++i) if (i != i0 && pr[i] > pr[i1]) i1 = i;
    float w0 = pr[i0], w1 = pr[i1];
    float inv = 1.f / (w0 + w1);       // softmax denominator cancels
    topk_idx[t * 2 + 0] = i0;
    topk_idx[t * 2 + 1] = i1;
    topk_w[t * 2 + 0] = w0 * inv;
    topk_w[t * 2 + 1] = w1 * inv;
    atomicAdd(&counts[i0], 1);
    atomicAdd(&counts[i1], 1);
  }
}

__global__ void moe_zero(int* __restrict__ counts) {
  if (threadIdx.x < EDIM) counts[threadIdx.x] = 0;
}

__global__ void moe_scan(const int* __restrict__ counts,
                         int* __restrict__ offsets,   // E+1
                         int* __restrict__ cursors)   // E
{
  if (threadIdx.x == 0) {
    int acc = 0;
    for (int e = 0; e < EDIM; ++e) { offsets[e] = acc; cursors[e] = acc; acc += counts[e]; }
    offsets[EDIM] = acc;               // == 2*T
  }
}

__global__ void moe_scatter(const int* __restrict__ topk_idx,
                            int* __restrict__ cursors,
                            int* __restrict__ tokens,   // 2*T
                            int* __restrict__ slot_of)  // T*2
{
  int t = blockIdx.x * blockDim.x + threadIdx.x;
  if (t >= T_TOK) return;
  #pragma unroll
  for (int k = 0; k < 2; ++k) {
    int e = topk_idx[t * 2 + k];
    int pos = atomicAdd(&cursors[e], 1);
    tokens[pos] = t;
    slot_of[t * 2 + k] = pos;
  }
}

// ---------------------------------------------------------------------------
// WMMA fragment maps (ISA 7.12.2, 16-bit operands, wave32), pair-packed u32:
//  A 16x32: lane<16 -> M=lane, elem j -> K=j; lane>=16 -> K+16 (contiguous order used here).
//           LDS A_s[m][q] u32 holds K=2q (lo),2q+1 (hi): fragment = two
//           consecutive i32x4 at word m*16 + (lane>=16?8:0) and +4.
//  B 32x16 (contiguous K order used here): lane<16 -> N=lane, K=j; lane>=16 -> N=lane-16, K=16+j.
//           LDS B_s[p][n] u32 holds K=2p,2p+1 at col n: fragment VGPR v =
//           word (p0+v)*128 + n, p0 = (lane>=16?8:0).
//  C 16x16 f32: v -> M = v + (lane>=16?8:0), N = lane&15.
// ---------------------------------------------------------------------------

// 4. Fused gate+up GEMM: h[slot, f] = silu(x@Wg) * (x@Wu), bf16 out.
//    grid = (E*MAX_RT, F/128), block = 256 (8 waves, 16 cols each).
__global__ void __launch_bounds__(256)
moe_gemm_gate_up(const float* __restrict__ x,   // T x D
                 const float* __restrict__ Wg,  // E x D x F
                 const float* __restrict__ Wu,  // E x D x F
                 const int*   __restrict__ offsets,
                 const int*   __restrict__ tokens,
                 bf16_t*      __restrict__ h_buf) // 2T x F
{
  const int e    = blockIdx.x >> 6;
  const int rt   = blockIdx.x & (MAX_RT - 1);
  const int row0 = offsets[e];
  const int rows = offsets[e + 1] - row0;
  if (rt * 16 >= rows) return;                       // uniform early exit

  const int tid  = threadIdx.x;
  const int lane = tid & 31;
  const int wave = tid >> 5;
  const int f0   = blockIdx.y * 128;

  __shared__ unsigned int A_s[16 * 16];    // 1KB  [m][kpair]
  __shared__ unsigned int Bg_s[16 * 128];  // 8KB  [kpair][n]
  __shared__ unsigned int Bu_s[16 * 128];  // 8KB

  // --- A staging: thread -> (row m = tid>>4, pair q = tid&15) ---
  const int am = tid >> 4;
  const int aq = tid & 15;
  int ar = rt * 16 + am; if (ar >= rows) ar = rows - 1;   // clamp padding
  const float* xrow = x + (size_t)tokens[row0 + ar] * DDIM + 2 * aq;

  // --- B staging: two tasks, rows 2*p0..2*p0+1 and +16, cols c..c+3 ---
  const int p0 = tid >> 5;              // 0..7
  const int c  = (tid & 31) << 2;       // 0..124
  const float* gp = Wg + (size_t)e * DDIM * FDIM + (size_t)(2 * p0) * FDIM + f0 + c;
  const float* up = Wu + (size_t)e * DDIM * FDIM + (size_t)(2 * p0) * FDIM + f0 + c;

  // fragment read addressing (constant across K)
  const int abase = (lane & 15) * 16 + ((lane >= 16) ? 8 : 0);
  const int nb    = wave * 16 + (lane & 15);
  const int pb    = (lane >= 16) ? 8 : 0;

  v8f accg = {}; v8f accu = {};

  // register-pipelined weight slab (8 x b128 per thread)
  f32x4 Rg0l, Rg0h, Ru0l, Ru0h, Rg1l, Rg1h, Ru1l, Ru1h;
  auto load_slab = [&](int k0) {
    const size_t b = (size_t)k0 * FDIM;
    Rg0l = *(const f32x4*)(gp + b);
    Rg0h = *(const f32x4*)(gp + b + FDIM);
    Ru0l = *(const f32x4*)(up + b);
    Ru0h = *(const f32x4*)(up + b + FDIM);
    Rg1l = *(const f32x4*)(gp + b + 16 * FDIM);
    Rg1h = *(const f32x4*)(gp + b + 17 * FDIM);
    Ru1l = *(const f32x4*)(up + b + 16 * FDIM);
    Ru1h = *(const f32x4*)(up + b + 17 * FDIM);
  };
  load_slab(0);

  for (int k0 = 0; k0 < DDIM; k0 += 32) {
    f32x2 av = *(const f32x2*)(xrow + k0);           // tiny A slab (b64)
    __syncthreads();                                  // LDS free
    A_s[am * 16 + aq] = pack2(av[0], av[1]);
    {
      i32x4 w0, w1, w2, w3;
      #pragma unroll
      for (int i = 0; i < 4; ++i) {
        w0[i] = (int)pack2(Rg0l[i], Rg0h[i]);
        w1[i] = (int)pack2(Ru0l[i], Ru0h[i]);
        w2[i] = (int)pack2(Rg1l[i], Rg1h[i]);
        w3[i] = (int)pack2(Ru1l[i], Ru1h[i]);
      }
      *(i32x4*)&Bg_s[p0 * 128 + c]       = w0;
      *(i32x4*)&Bu_s[p0 * 128 + c]       = w1;
      *(i32x4*)&Bg_s[(p0 + 8) * 128 + c] = w2;
      *(i32x4*)&Bu_s[(p0 + 8) * 128 + c] = w3;
    }
    __syncthreads();                                  // LDS ready

    // prefetch next slab into registers (overlaps frag reads + WMMA)
    if (k0 + 32 < DDIM) load_slab(k0 + 32);

    // A fragment: two 16B LDS reads
    i32x4 a0 = *(const i32x4*)&A_s[abase];
    i32x4 a1 = *(const i32x4*)&A_s[abase + 4];
    i32x8 aw;
    #pragma unroll
    for (int i = 0; i < 4; ++i) { aw[i] = a0[i]; aw[i + 4] = a1[i]; }
    v16bf af = __builtin_bit_cast(v16bf, aw);

    // B fragments: 8 x b32 LDS reads each (stride 512B, conflict-free)
    i32x8 gwv, uwv;
    #pragma unroll
    for (int v = 0; v < 8; ++v) {
      gwv[v] = (int)Bg_s[(pb + v) * 128 + nb];
      uwv[v] = (int)Bu_s[(pb + v) * 128 + nb];
    }
    v16bf bg = __builtin_bit_cast(v16bf, gwv);
    v16bf bu = __builtin_bit_cast(v16bf, uwv);

    accg = __builtin_amdgcn_wmma_f32_16x16x32_bf16(false, af, false, bg,
                                                   (short)0, accg, false, false);
    accu = __builtin_amdgcn_wmma_f32_16x16x32_bf16(false, af, false, bu,
                                                   (short)0, accu, false, false);
  }

  // epilogue: silu(g) * u -> bf16 h rows
  const int n = f0 + nb;
  #pragma unroll
  for (int v = 0; v < 8; ++v) {
    int r = rt * 16 + v + ((lane >= 16) ? 8 : 0);
    if (r < rows) {
      float g = accg[v];
      float s = g / (1.0f + __expf(-g));
      h_buf[(size_t)(row0 + r) * FDIM + n] = (__bf16)(s * accu[v]);
    }
  }
}

// 5. Down GEMM: down[slot, d] = h[slot, :] @ Wd[e].  K = F = 2048.
//    grid = (E*MAX_RT, D/128), block = 256.  A read directly from global
//    (h is already bf16): two global_load_b128 per lane per K-slab.
__global__ void __launch_bounds__(256)
moe_gemm_down(const bf16_t* __restrict__ h_buf, // 2T x F (bf16)
              const float*  __restrict__ Wd,    // E x F x D
              const int*    __restrict__ offsets,
              float*        __restrict__ down_buf) // 2T x D
{
  const int e    = blockIdx.x >> 6;
  const int rt   = blockIdx.x & (MAX_RT - 1);
  const int row0 = offsets[e];
  const int rows = offsets[e + 1] - row0;
  if (rt * 16 >= rows) return;

  const int tid  = threadIdx.x;
  const int lane = tid & 31;
  const int wave = tid >> 5;
  const int d0   = blockIdx.y * 128;

  __shared__ unsigned int Bd_s[16 * 128];  // 8KB [kpair][n]

  // A: per-lane row pointer (hoisted), K chunk offset baked in
  const int mA = lane & 15;
  int ar = rt * 16 + mA; if (ar >= rows) ar = rows - 1;
  const bf16_t* hrow = h_buf + (size_t)(row0 + ar) * FDIM + ((lane >= 16) ? 16 : 0);

  const int p0 = tid >> 5;
  const int c  = (tid & 31) << 2;
  const float* dp = Wd + (size_t)e * FDIM * DDIM + (size_t)(2 * p0) * DDIM + d0 + c;

  const int nb = wave * 16 + (lane & 15);
  const int pb = (lane >= 16) ? 8 : 0;

  v8f acc = {};

  f32x4 R0l, R0h, R1l, R1h;
  auto load_slab = [&](int k0) {
    const size_t b = (size_t)k0 * DDIM;
    R0l = *(const f32x4*)(dp + b);
    R0h = *(const f32x4*)(dp + b + DDIM);
    R1l = *(const f32x4*)(dp + b + 16 * DDIM);
    R1h = *(const f32x4*)(dp + b + 17 * DDIM);
  };
  load_slab(0);

  for (int k0 = 0; k0 < FDIM; k0 += 32) {
    // A fragment loads issued early (global, consumed after barrier)
    i32x4 a0 = *(const i32x4*)(hrow + k0);
    i32x4 a1 = *(const i32x4*)(hrow + k0 + 8);

    __syncthreads();
    {
      i32x4 w0, w1;
      #pragma unroll
      for (int i = 0; i < 4; ++i) {
        w0[i] = (int)pack2(R0l[i], R0h[i]);
        w1[i] = (int)pack2(R1l[i], R1h[i]);
      }
      *(i32x4*)&Bd_s[p0 * 128 + c]       = w0;
      *(i32x4*)&Bd_s[(p0 + 8) * 128 + c] = w1;
    }
    __syncthreads();

    if (k0 + 32 < FDIM) load_slab(k0 + 32);   // overlap with WMMA

    i32x8 aw;
    #pragma unroll
    for (int i = 0; i < 4; ++i) { aw[i] = a0[i]; aw[i + 4] = a1[i]; }
    v16bf af = __builtin_bit_cast(v16bf, aw);

    i32x8 bwv;
    #pragma unroll
    for (int v = 0; v < 8; ++v) bwv[v] = (int)Bd_s[(pb + v) * 128 + nb];
    v16bf bd = __builtin_bit_cast(v16bf, bwv);

    acc = __builtin_amdgcn_wmma_f32_16x16x32_bf16(false, af, false, bd,
                                                  (short)0, acc, false, false);
  }

  const int n = d0 + nb;
  #pragma unroll
  for (int v = 0; v < 8; ++v) {
    int r = rt * 16 + v + ((lane >= 16) ? 8 : 0);
    if (r < rows) down_buf[(size_t)(row0 + r) * DDIM + n] = acc[v];
  }
}

// 6. Combine (vectorized f32x4): out[t,d] = w0*down[s0,d] + w1*down[s1,d]
__global__ void moe_combine(const float* __restrict__ down_buf,
                            const float* __restrict__ topk_w,
                            const int*   __restrict__ slot_of,
                            float*       __restrict__ out)
{
  int i  = blockIdx.x * blockDim.x + threadIdx.x;  // over T*D/4
  int t  = i >> 8;                                  // D/4 = 256 quads/token
  int dq = (i & 255) << 2;
  float w0 = topk_w[t * 2 + 0], w1 = topk_w[t * 2 + 1];
  int   s0 = slot_of[t * 2 + 0], s1 = slot_of[t * 2 + 1];
  f32x4 a = *(const f32x4*)(down_buf + (size_t)s0 * DDIM + dq);
  f32x4 b = *(const f32x4*)(down_buf + (size_t)s1 * DDIM + dq);
  f32x4 o;
  #pragma unroll
  for (int j = 0; j < 4; ++j) o[j] = w0 * a[j] + w1 * b[j];
  *(f32x4*)(out + (size_t)t * DDIM + dq) = o;
}

// ---------------------------------------------------------------------------
// Launch.  Workspace layout (bytes):
//   counts   @ 0        (64)
//   offsets  @ 256      (68)
//   cursors  @ 512      (64)
//   topk_idx @ 1024     (8K)
//   slot_of  @ 9216     (8K)
//   topk_w   @ 17408    (8K)
//   tokens   @ 25600    (8K)
//   h_buf    @ 33792    (2T*F bf16 = 8 MB)
//   down_buf @ 8422400  (2T*D f32 = 8 MB)     total ~16.05 MB
// ---------------------------------------------------------------------------
extern "C" void kernel_launch(void* const* d_in, const int* in_sizes, int n_in,
                              void* d_out, int out_size, void* d_ws, size_t ws_size,
                              hipStream_t stream) {
  const float* x  = (const float*)d_in[0];
  const float* Wr = (const float*)d_in[1];
  const float* Wg = (const float*)d_in[2];
  const float* Wu = (const float*)d_in[3];
  const float* Wd = (const float*)d_in[4];
  float* out = (float*)d_out;

  char* ws = (char*)d_ws;
  int*    counts   = (int*)(ws + 0);
  int*    offsets  = (int*)(ws + 256);
  int*    cursors  = (int*)(ws + 512);
  int*    topk_idx = (int*)(ws + 1024);
  int*    slot_of  = (int*)(ws + 9216);
  float*  topk_w   = (float*)(ws + 17408);
  int*    tokens   = (int*)(ws + 25600);
  bf16_t* h_buf    = (bf16_t*)(ws + 33792);
  float*  down_buf = (float*)(ws + 8422400);

  moe_zero<<<1, 32, 0, stream>>>(counts);
  moe_router<<<T_TOK, 256, 0, stream>>>(x, Wr, topk_w, topk_idx, counts);
  moe_scan<<<1, 32, 0, stream>>>(counts, offsets, cursors);
  moe_scatter<<<(T_TOK + 255) / 256, 256, 0, stream>>>(topk_idx, cursors, tokens, slot_of);
  moe_gemm_gate_up<<<dim3(EDIM * MAX_RT, FDIM / 128), 256, 0, stream>>>(
      x, Wg, Wu, offsets, tokens, h_buf);
  moe_gemm_down<<<dim3(EDIM * MAX_RT, DDIM / 128), 256, 0, stream>>>(
      h_buf, Wd, offsets, down_buf);
  moe_combine<<<(T_TOK * DDIM) / 4 / 256, 256, 0, stream>>>(down_buf, topk_w, slot_of, out);
}
